// NeuralODESolver_16793322127505
// MI455X (gfx1250) — matrix operand order repaired and checked by reading, not compiled, not run on hardware
//
#include <hip/hip_runtime.h>
#include <hip/hip_bf16.h>

typedef __attribute__((ext_vector_type(16))) _Float16 v16h;
typedef __attribute__((ext_vector_type(8)))  _Float16 v8h;
typedef __attribute__((ext_vector_type(8)))  float    v8f;

namespace {

constexpr int Bn  = 1024;   // batch
constexpr int Dn  = 64;     // state dim
constexpr int Un  = 64;     // input dim
constexpr int DUn = 128;    // D+U
constexpr int Hn  = 256;    // hidden
constexpr int MT  = 16;     // batch rows per workgroup
constexpr int NW  = 4;      // waves per workgroup (wave32)
constexpr int ZLD = DUn + 8; // padded LDS row (f16) -> +16B bank skew
constexpr int ALD = Hn + 8;

// Tsitouras 5(4), premultiplied by h = 60 s = 1/60 hr
constexpr double HS = 60.0 / 3600.0;
constexpr float hA21 = (float)(HS *  0.161);
constexpr float hA31 = (float)(HS * -0.008480655492356989);
constexpr float hA32 = (float)(HS *  0.335480655492357);
constexpr float hA41 = (float)(HS *  2.8971530571054935);
constexpr float hA42 = (float)(HS * -6.359448489975075);
constexpr float hA43 = (float)(HS *  4.3622954328695815);
constexpr float hA51 = (float)(HS *  5.325864828439257);
constexpr float hA52 = (float)(HS * -11.748883564062828);
constexpr float hA53 = (float)(HS *  7.4955393428898365);
constexpr float hA54 = (float)(HS * -0.09249506636175525);
constexpr float hA61 = (float)(HS *  5.86145544294642);
constexpr float hA62 = (float)(HS * -12.92096931784711);
constexpr float hA63 = (float)(HS *  8.159367898576159);
constexpr float hA64 = (float)(HS * -0.071584973281401);
constexpr float hA65 = (float)(HS * -0.028269050394068383);
constexpr float hB1  = (float)(HS *  0.09646076681806523);
constexpr float hB2  = (float)(HS *  0.01);
constexpr float hB3  = (float)(HS *  0.4798896504144996);
constexpr float hB4  = (float)(HS *  1.379008574103742);
constexpr float hB5  = (float)(HS * -3.290069515436081);
constexpr float hB6  = (float)(HS *  2.324710524099774);

// packed-weight offsets in workspace (f16 units)
constexpr int W1T = 0;                 // (128/32)*(256/16)*512 = 32768
constexpr int W2T = 32768;             // (256/32)*(256/16)*512 = 65536
constexpr int W3T = 32768 + 65536;     // (256/32)*( 64/16)*512 = 16384

} // namespace

// ---------------------------------------------------------------------------
// Pack fp32 weight W[Nout][Kin] (row-major) into WMMA B-fragment tiles of f16.
// Tile (kt,nt) covers K = kt*32..+31, N = nt*16..+15.  Within a tile, lane L
// holds column n = L%16 for K-group g = L/16; element r (0..15) of the lane's
// v16h is k_local = r + 16*g.  B(k,n) = W[n][k]  (out = act @ W^T).
// ---------------------------------------------------------------------------
__global__ void pack_w_kernel(const float* __restrict__ W,
                              _Float16* __restrict__ dst,
                              int Ktot, int Ntot) {
  int tid = blockIdx.x * blockDim.x + threadIdx.x;
  int total = Ktot * Ntot;
  if (tid >= total) return;
  int ntiles = Ntot >> 4;
  int tile = tid >> 9;          // /512 elements per tile
  int p    = tid & 511;
  int L    = p >> 4;            // lane 0..31
  int r    = p & 15;            // element within lane
  int kt   = tile / ntiles;
  int nt   = tile % ntiles;
  int g    = L >> 4;
  int nl   = L & 15;
  int k    = kt * 32 + r + 16 * g;
  int n    = nt * 16 + nl;
  dst[tid] = (_Float16)W[(size_t)n * Ktot + k];
}

// Build a 16x32 f16 A-fragment for lane (ln, g) from a row-major LDS row.
// Layout: lane group g covers k = 16g..16g+15 (elems 0-15), matching packed B.
__device__ __forceinline__ v16h load_afrag(const _Float16* rowp, int g) {
  v8h lo = *(const v8h*)(rowp + 16 * g);
  v8h hi = *(const v8h*)(rowp + 16 * g + 8);
  return __builtin_shufflevector(lo, hi, 0, 1, 2, 3, 4, 5, 6, 7,
                                         8, 9, 10, 11, 12, 13, 14, 15);
}

// One MLP evaluation: kout = W3 @ relu(W2 @ relu(W1 @ [yin|u] + b1) + b2) + b3
// yin/kout live in the WMMA C-fragment layout:
//   element v of this thread = (row m = v + 8g, col n = nq*16 + ln)
__device__ __forceinline__ void mlp_eval(
    const float* yin, float* kout,
    _Float16 (&z)[MT][ZLD], _Float16 (&a)[MT][ALD],
    const _Float16* __restrict__ W1p,
    const _Float16* __restrict__ W2p,
    const _Float16* __restrict__ W3p,
    const float* b1r, const float* b2r, float b3r,
    int nq, int g, int ln, int lane)
{
  // scatter stage input (f32 -> f16) into state columns of z = [y | u]
#pragma unroll
  for (int v = 0; v < 8; ++v)
    z[v + 8 * g][nq * 16 + ln] = (_Float16)yin[v];
  __syncthreads();

  // ---- layer 1: [16,128] x [128,256] -> a1 ----
  v8f acc[4] = {};
#pragma unroll
  for (int kt = 0; kt < DUn / 32; ++kt) {
    v16h af = load_afrag(&z[ln][kt * 32], g);
#pragma unroll
    for (int j = 0; j < 4; ++j) {
      v16h bf = *(const v16h*)(W1p + (size_t)(kt * 16 + nq * 4 + j) * 512 + lane * 16);
      acc[j] = __builtin_amdgcn_wmma_f32_16x16x32_f16(
          false, af, false, bf, (short)0, acc[j], false, false);
    }
  }
  __syncthreads();
#pragma unroll
  for (int j = 0; j < 4; ++j)
#pragma unroll
    for (int v = 0; v < 8; ++v) {
      float t = acc[j][v] + b1r[j];
      a[v + 8 * g][(nq * 4 + j) * 16 + ln] = (_Float16)(t > 0.0f ? t : 0.0f);
    }
  __syncthreads();

  // ---- layer 2: [16,256] x [256,256] -> a2 (in-place over a1) ----
#pragma unroll
  for (int j = 0; j < 4; ++j) acc[j] = (v8f){};
#pragma unroll
  for (int kt = 0; kt < Hn / 32; ++kt) {
    v16h af = load_afrag(&a[ln][kt * 32], g);
#pragma unroll
    for (int j = 0; j < 4; ++j) {
      v16h bf = *(const v16h*)(W2p + (size_t)(kt * 16 + nq * 4 + j) * 512 + lane * 16);
      acc[j] = __builtin_amdgcn_wmma_f32_16x16x32_f16(
          false, af, false, bf, (short)0, acc[j], false, false);
    }
  }
  __syncthreads();  // all a1 reads done before overwrite
#pragma unroll
  for (int j = 0; j < 4; ++j)
#pragma unroll
    for (int v = 0; v < 8; ++v) {
      float t = acc[j][v] + b2r[j];
      a[v + 8 * g][(nq * 4 + j) * 16 + ln] = (_Float16)(t > 0.0f ? t : 0.0f);
    }
  __syncthreads();

  // ---- layer 3: [16,256] x [256,64] -> kout (register fragment) ----
  v8f c3 = {};
#pragma unroll
  for (int kt = 0; kt < Hn / 32; ++kt) {
    v16h af = load_afrag(&a[ln][kt * 32], g);
    v16h bf = *(const v16h*)(W3p + (size_t)(kt * 4 + nq) * 512 + lane * 16);
    c3 = __builtin_amdgcn_wmma_f32_16x16x32_f16(
        false, af, false, bf, (short)0, c3, false, false);
  }
#pragma unroll
  for (int v = 0; v < 8; ++v) kout[v] = c3[v] + b3r;
}

__global__ __launch_bounds__(MT * NW * 2) void ode_rk5_kernel(
    const float* __restrict__ x0, const float* __restrict__ u,
    const float* __restrict__ b1, const float* __restrict__ b2,
    const float* __restrict__ b3,
    const int* __restrict__ t0p, const int* __restrict__ t1p,
    const _Float16* __restrict__ wpack, float* __restrict__ out)
{
  __shared__ _Float16 z[MT][ZLD];   // [y | u] as f16, row-major, padded
  __shared__ _Float16 a[MT][ALD];   // hidden activations

  const int lane = threadIdx.x & 31;
  const int nq   = threadIdx.x >> 5;  // wave id = column quarter
  const int g    = lane >> 4;
  const int ln   = lane & 15;
  const int rowbase = blockIdx.x * MT;
  const int nn = nq * 16 + ln;        // owned output column (0..63)

  const _Float16* W1p = wpack + W1T;
  const _Float16* W2p = wpack + W2T;
  const _Float16* W3p = wpack + W3T;

  // Load initial state into C-fragment registers; stage constant u into z.
  float y[8];
#pragma unroll
  for (int v = 0; v < 8; ++v) {
    int m = v + 8 * g;
    y[v] = x0[(size_t)(rowbase + m) * Dn + nn];
    z[m][Dn + nn] = (_Float16)u[(size_t)(rowbase + m) * Un + nn];
  }
  float b1r[4], b2r[4];
#pragma unroll
  for (int j = 0; j < 4; ++j) {
    b1r[j] = b1[(nq * 4 + j) * 16 + ln];
    b2r[j] = b2[(nq * 4 + j) * 16 + ln];
  }
  const float b3r = b3[nn];

  const int nsteps = 60 * (t1p[0] - t0p[0]);

  float k1[8], k2[8], k3[8], k4[8], k5[8], k6[8], tmp[8];
  for (int s = 0; s < nsteps; ++s) {
    mlp_eval(y, k1, z, a, W1p, W2p, W3p, b1r, b2r, b3r, nq, g, ln, lane);
#pragma unroll
    for (int v = 0; v < 8; ++v)
      tmp[v] = fmaf(hA21, k1[v], y[v]);
    mlp_eval(tmp, k2, z, a, W1p, W2p, W3p, b1r, b2r, b3r, nq, g, ln, lane);
#pragma unroll
    for (int v = 0; v < 8; ++v)
      tmp[v] = fmaf(hA32, k2[v], fmaf(hA31, k1[v], y[v]));
    mlp_eval(tmp, k3, z, a, W1p, W2p, W3p, b1r, b2r, b3r, nq, g, ln, lane);
#pragma unroll
    for (int v = 0; v < 8; ++v)
      tmp[v] = fmaf(hA43, k3[v], fmaf(hA42, k2[v], fmaf(hA41, k1[v], y[v])));
    mlp_eval(tmp, k4, z, a, W1p, W2p, W3p, b1r, b2r, b3r, nq, g, ln, lane);
#pragma unroll
    for (int v = 0; v < 8; ++v)
      tmp[v] = fmaf(hA54, k4[v],
               fmaf(hA53, k3[v], fmaf(hA52, k2[v], fmaf(hA51, k1[v], y[v]))));
    mlp_eval(tmp, k5, z, a, W1p, W2p, W3p, b1r, b2r, b3r, nq, g, ln, lane);
#pragma unroll
    for (int v = 0; v < 8; ++v)
      tmp[v] = fmaf(hA65, k5[v], fmaf(hA64, k4[v],
               fmaf(hA63, k3[v], fmaf(hA62, k2[v], fmaf(hA61, k1[v], y[v])))));
    mlp_eval(tmp, k6, z, a, W1p, W2p, W3p, b1r, b2r, b3r, nq, g, ln, lane);
#pragma unroll
    for (int v = 0; v < 8; ++v)
      y[v] = fmaf(hB6, k6[v], fmaf(hB5, k5[v], fmaf(hB4, k4[v],
             fmaf(hB3, k3[v], fmaf(hB2, k2[v], fmaf(hB1, k1[v], y[v]))))));
  }

#pragma unroll
  for (int v = 0; v < 8; ++v)
    out[(size_t)(rowbase + v + 8 * g) * Dn + nn] = y[v];
}

extern "C" void kernel_launch(void* const* d_in, const int* in_sizes, int n_in,
                              void* d_out, int out_size, void* d_ws, size_t ws_size,
                              hipStream_t stream) {
  (void)in_sizes; (void)n_in; (void)out_size; (void)ws_size;
  const float* x0 = (const float*)d_in[0];
  const float* u  = (const float*)d_in[1];
  const float* W1 = (const float*)d_in[2];
  const float* b1 = (const float*)d_in[3];
  const float* W2 = (const float*)d_in[4];
  const float* b2 = (const float*)d_in[5];
  const float* W3 = (const float*)d_in[6];
  const float* b3 = (const float*)d_in[7];
  const int*   t0 = (const int*)d_in[8];
  const int*   t1 = (const int*)d_in[9];
  _Float16* wp = (_Float16*)d_ws;

  // Pack weights (fp32 row-major -> f16 WMMA B fragments) each call; cheap and
  // deterministic. Total packed size = 224 KB of workspace.
  pack_w_kernel<<<(DUn * Hn) / 256, 256, 0, stream>>>(W1, wp + W1T, DUn, Hn);
  pack_w_kernel<<<(Hn * Hn)  / 256, 256, 0, stream>>>(W2, wp + W2T, Hn, Hn);
  pack_w_kernel<<<(Hn * Dn)  / 256, 256, 0, stream>>>(W3, wp + W3T, Hn, Dn);

  ode_rk5_kernel<<<Bn / MT, MT * NW * 2, 0, stream>>>(
      x0, u, b1, b2, b3, t0, t1, wp, (float*)d_out);
}
